// Attention_15968688406887
// MI455X (gfx1250) — matrix operand order repaired and checked by reading, not compiled, not run on hardware
//
#include <hip/hip_runtime.h>
#include <hip/hip_bf16.h>

// ---------------------------------------------------------------------------
// MHA for MI455X (gfx1250, wave32) via V_WMMA_F32_16X16X32_BF16.
//   pack:  fp32 -> bf16 one-shot conversion of input/memory/weights (ws)
//   proj:  Q = (Xb @ WqbT + bq) * 0.125  -> bf16 [B,H,L,64]   (ws)
//          K = (Mb @ WkbT + bk)          -> bf16 [B,H,L,64]   (ws)
//          Vt= (Mb @ WvbT + bv)^T        -> bf16 [B,H,64,L]   (ws)
//   attn:  flash-style online softmax (64-key chunks), out fp32 [B,N,1024]
// Workspace: 5*8192*1024 + 3*1024*1024 bf16 ~= 90 MB of d_ws.
// ---------------------------------------------------------------------------

typedef __attribute__((ext_vector_type(16))) __bf16 v16bf;
typedef __attribute__((ext_vector_type(8)))  __bf16 v8bf;
typedef __attribute__((ext_vector_type(8)))  float  v8f;

#define DIMN   1024
#define SEQL   2048
#define BATCH  4
#define NHEAD  16
#define HDIM   64
#define ROWS   (BATCH * SEQL)   // 8192

__device__ __forceinline__ v8f wmma_bf16(v16bf a, v16bf b, v8f c) {
    // D = A(16x32 bf16) x B(32x16 bf16) + C(16x16 f32)
    return __builtin_amdgcn_wmma_f32_16x16x32_bf16(
        /*neg_a=*/false, a, /*neg_b=*/false, b,
        /*c_mod=*/(short)0, c, /*reuse_a=*/false, /*reuse_b=*/false);
}

// ---------------------------------------------------------------------------
// Elementwise fp32 -> bf16 pack (8 elements / thread, b128 in, b128 out)
// ---------------------------------------------------------------------------
__global__ void __launch_bounds__(256)
pack_bf16_kernel(const float* __restrict__ src, __bf16* __restrict__ dst, int n8)
{
    int i = blockIdx.x * 256 + threadIdx.x;
    if (i < n8) {
        const float4* s = (const float4*)src + (size_t)i * 2;
        float4 a = s[0], b = s[1];
        v8bf p;
        p[0] = (__bf16)a.x; p[1] = (__bf16)a.y; p[2] = (__bf16)a.z; p[3] = (__bf16)a.w;
        p[4] = (__bf16)b.x; p[5] = (__bf16)b.y; p[6] = (__bf16)b.z; p[7] = (__bf16)b.w;
        ((v8bf*)dst)[i] = p;
    }
}

// ---------------------------------------------------------------------------
// Projection: Y[r,n] = (sum_k X[r,k] * W[n,k] + bias[n]) * scale, X/W bf16.
// One 16x64 output strip per wave (4 WMMAs / k-step, A-fragment reused 4x,
// zero conversions in the k-loop). 8 waves / block.
// Output layout: head-split bf16. transposeOut=0 -> [B,H,L,64]
//                                 transposeOut=1 -> [B,H,64,L]
// ---------------------------------------------------------------------------
__global__ void __launch_bounds__(256)
proj_kernel(const __bf16* __restrict__ X, const __bf16* __restrict__ W,
            const float* __restrict__ bias, __bf16* __restrict__ Y,
            float scale, int transposeOut)
{
    const int lane    = threadIdx.x & 31;
    const int wave    = threadIdx.x >> 5;
    const int half    = (lane < 16) ? 0 : 1;
    const int l16     = lane & 15;
    const int rowBase = blockIdx.x * 16;
    const int colBase = (blockIdx.y * 8 + wave) * 64;   // 64-wide strip: one head

    const int arow = rowBase + l16;                     // A-fragment row (M)
    const __bf16* xrow = X + (size_t)arow * DIMN;

    const __bf16* wrow[4];
    #pragma unroll
    for (int c = 0; c < 4; ++c)
        wrow[c] = W + (size_t)(colBase + c * 16 + l16) * DIMN;  // B-frag col (N)

    const int koffA = half * 16;    // A: lanes 0-15 hold k 0-15, hi 16-31
    const int koffB = half * 16;    // B: lanes 0-15 hold k 0-15, hi 16-31

    v8f acc[4] = {{}, {}, {}, {}};

    for (int kb = 0; kb < DIMN; kb += 32) {
        v8bf alo = *(const v8bf*)(xrow + kb + koffA);
        v8bf ahi = *(const v8bf*)(xrow + kb + koffA + 8);
        v16bf afrag;
        #pragma unroll
        for (int i = 0; i < 8; ++i) { afrag[i] = alo[i]; afrag[i + 8] = ahi[i]; }

        #pragma unroll
        for (int c = 0; c < 4; ++c) {
            v16bf bfrag = *(const v16bf*)(wrow[c] + kb + koffB);  // 32 B contiguous
            acc[c] = wmma_bf16(afrag, bfrag, acc[c]);
        }
    }

    const int rowOff = half * 8;                 // C: vgpr r is row r (lo) / r+8 (hi)
    const int bidx   = rowBase / SEQL;
    const int h      = colBase / HDIM;           // whole strip is one head
    const int lbase  = (rowBase % SEQL) + rowOff;

    if (!transposeOut) {
        __bf16* yb = Y + (size_t)(bidx * NHEAD + h) * SEQL * HDIM;
        #pragma unroll
        for (int c = 0; c < 4; ++c) {
            const int d    = c * 16 + l16;
            const float bn = bias[colBase + d];
            #pragma unroll
            for (int r = 0; r < 8; ++r) {
                float v = (acc[c][r] + bn) * scale;
                yb[(size_t)(lbase + r) * HDIM + d] = (__bf16)v;
            }
        }
    } else {
        #pragma unroll
        for (int c = 0; c < 4; ++c) {
            const int d    = c * 16 + l16;
            const float bn = bias[colBase + d];
            v8bf pack;
            #pragma unroll
            for (int r = 0; r < 8; ++r)
                pack[r] = (__bf16)((acc[c][r] + bn) * scale);
            __bf16* yb = Y + ((size_t)(bidx * NHEAD + h) * HDIM + d) * SEQL + lbase;
            *(v8bf*)yb = pack;   // 8 consecutive L positions -> one b128 store
        }
    }
}

// ---------------------------------------------------------------------------
// Flash attention: one wave handles a 16-query tile of one (b,h); streams
// keys/values in chunks of 64 with online softmax (16 WMMAs per chunk).
// ---------------------------------------------------------------------------
__global__ void __launch_bounds__(32)
attn_kernel(const __bf16* __restrict__ Q, const __bf16* __restrict__ K,
            const __bf16* __restrict__ Vt, float* __restrict__ out)
{
    __shared__ __bf16 Pl[16][64];        // P tile staged C-layout -> A-layout (2 KB)

    const int lane  = threadIdx.x & 31;
    const int half  = (lane < 16) ? 0 : 1;
    const int l16   = lane & 15;
    const int bh    = blockIdx.y;
    const int b     = bh >> 4;
    const int h     = bh & 15;
    const int qBase = blockIdx.x * 16;

    const __bf16* Qbh = Q  + (size_t)bh * SEQL * HDIM;
    const __bf16* Kbh = K  + (size_t)bh * SEQL * HDIM;
    const __bf16* Vbh = Vt + (size_t)bh * HDIM * SEQL;

    const int koffA = half * 16;
    const int koffB = half * 16;

    // Preload Q A-fragments for d = 0..63 (two 16x32 fragments).
    v16bf qa[2];
    #pragma unroll
    for (int dstep = 0; dstep < 2; ++dstep) {
        const __bf16* p = Qbh + (size_t)(qBase + l16) * HDIM + dstep * 32 + koffA;
        v8bf lo = *(const v8bf*)(p);
        v8bf hi = *(const v8bf*)(p + 8);
        #pragma unroll
        for (int i = 0; i < 8; ++i) { qa[dstep][i] = lo[i]; qa[dstep][i + 8] = hi[i]; }
    }

    v8f o[4] = {{}, {}, {}, {}};         // 16x64 fp32 output accumulator
    float m[8], lsum[8];
    #pragma unroll
    for (int r = 0; r < 8; ++r) { m[r] = -3.0e38f; lsum[r] = 0.0f; }

    for (int jb = 0; jb < SEQL; jb += 64) {
        // ---- S = Q . K^T for 64 keys (four 16x16 logit tiles, 8 WMMAs) ----
        v8f s[4] = {{}, {}, {}, {}};
        #pragma unroll
        for (int dstep = 0; dstep < 2; ++dstep) {
            #pragma unroll
            for (int g = 0; g < 4; ++g) {
                v16bf kf = *(const v16bf*)(
                    Kbh + (size_t)(jb + g * 16 + l16) * HDIM + dstep * 32 + koffB);
                s[g] = wmma_bf16(qa[dstep], kf, s[g]);
            }
        }

        // ---- issue V B-fragment loads early (overlap with softmax VALU) ----
        v16bf vf[4][2];
        #pragma unroll
        for (int c = 0; c < 4; ++c)
            #pragma unroll
            for (int kstep = 0; kstep < 2; ++kstep)
                vf[c][kstep] = *(const v16bf*)(
                    Vbh + (size_t)(c * 16 + l16) * SEQL + jb + kstep * 32 + koffB);

        // ---- online softmax (row stats via 16-lane xor reductions) ----
        float mnew[8], corr[8];
        #pragma unroll
        for (int r = 0; r < 8; ++r) {
            float v = fmaxf(fmaxf(s[0][r], s[1][r]), fmaxf(s[2][r], s[3][r]));
            v = fmaxf(v, __shfl_xor(v, 1, 32));
            v = fmaxf(v, __shfl_xor(v, 2, 32));
            v = fmaxf(v, __shfl_xor(v, 4, 32));
            v = fmaxf(v, __shfl_xor(v, 8, 32));
            mnew[r] = fmaxf(m[r], v);
            corr[r] = __expf(m[r] - mnew[r]);
            m[r]    = mnew[r];
        }
        float p[4][8];
        #pragma unroll
        for (int r = 0; r < 8; ++r) {
            float ps = 0.0f;
            #pragma unroll
            for (int g = 0; g < 4; ++g) {
                p[g][r] = __expf(s[g][r] - mnew[r]);
                ps += p[g][r];
            }
            ps += __shfl_xor(ps, 1, 32);
            ps += __shfl_xor(ps, 2, 32);
            ps += __shfl_xor(ps, 4, 32);
            ps += __shfl_xor(ps, 8, 32);
            lsum[r] = lsum[r] * corr[r] + ps;
        }
        #pragma unroll
        for (int c = 0; c < 4; ++c)
            #pragma unroll
            for (int r = 0; r < 8; ++r)
                o[c][r] *= corr[r];

        // ---- P (C-layout, f32) -> LDS bf16 -> two A-fragments ----
        #pragma unroll
        for (int g = 0; g < 4; ++g)
            #pragma unroll
            for (int r = 0; r < 8; ++r)
                Pl[half * 8 + r][g * 16 + l16] = (__bf16)p[g][r];
        __syncthreads();
        v16bf pa[2];
        #pragma unroll
        for (int kstep = 0; kstep < 2; ++kstep) {
            v8bf lo = *(const v8bf*)&Pl[l16][kstep * 32 + koffA];
            v8bf hi = *(const v8bf*)&Pl[l16][kstep * 32 + koffA + 8];
            #pragma unroll
            for (int i = 0; i < 8; ++i) { pa[kstep][i] = lo[i]; pa[kstep][i + 8] = hi[i]; }
        }
        __syncthreads();

        // ---- O += P . V  (Vt stored [64, L] -> contiguous B-fragments) ----
        #pragma unroll
        for (int c = 0; c < 4; ++c) {
            o[c] = wmma_bf16(pa[0], vf[c][0], o[c]);
            o[c] = wmma_bf16(pa[1], vf[c][1], o[c]);
        }
    }

    // ---- epilogue: normalize and store fp32 [B, N, H*64] ----
    #pragma unroll
    for (int r = 0; r < 8; ++r) {
        float inv = 1.0f / lsum[r];
        int row = qBase + half * 8 + r;
        float* op = out + ((size_t)b * SEQL + row) * DIMN + h * HDIM;
        #pragma unroll
        for (int c = 0; c < 4; ++c)
            op[c * 16 + l16] = o[c][r] * inv;
    }
}

extern "C" void kernel_launch(void* const* d_in, const int* in_sizes, int n_in,
                              void* d_out, int out_size, void* d_ws, size_t ws_size,
                              hipStream_t stream) {
    (void)in_sizes; (void)n_in; (void)out_size; (void)ws_size;
    const float* input  = (const float*)d_in[0];
    const float* memory = (const float*)d_in[1];
    const float* Wq     = (const float*)d_in[2];
    const float* bq     = (const float*)d_in[3];
    const float* Wk     = (const float*)d_in[4];
    const float* bk     = (const float*)d_in[5];
    const float* Wv     = (const float*)d_in[6];
    const float* bv     = (const float*)d_in[7];
    float* out          = (float*)d_out;

    const size_t NX = (size_t)ROWS * DIMN;            // 8.39M elements
    const size_t NW = (size_t)DIMN * DIMN;            // 1.05M elements

    __bf16* Qw  = (__bf16*)d_ws;
    __bf16* Kw  = Qw  + NX;
    __bf16* Vw  = Kw  + NX;
    __bf16* Xb  = Vw  + NX;
    __bf16* Mb  = Xb  + NX;
    __bf16* Wqb = Mb  + NX;
    __bf16* Wkb = Wqb + NW;
    __bf16* Wvb = Wkb + NW;

    // ---- fp32 -> bf16 packing (one-shot, removes all cvts from GEMM loop) ----
    {
        int n8x = (int)(NX / 8), n8w = (int)(NW / 8);
        pack_bf16_kernel<<<(n8x + 255) / 256, 256, 0, stream>>>(input,  Xb,  n8x);
        pack_bf16_kernel<<<(n8x + 255) / 256, 256, 0, stream>>>(memory, Mb,  n8x);
        pack_bf16_kernel<<<(n8w + 255) / 256, 256, 0, stream>>>(Wq,     Wqb, n8w);
        pack_bf16_kernel<<<(n8w + 255) / 256, 256, 0, stream>>>(Wk,     Wkb, n8w);
        pack_bf16_kernel<<<(n8w + 255) / 256, 256, 0, stream>>>(Wv,     Wvb, n8w);
    }

    dim3 pgrid(ROWS / 16, DIMN / (64 * 8));           // (512, 2)
    dim3 pblock(256);
    const float depth_scale = 0.125f;                 // 64^-0.5

    proj_kernel<<<pgrid, pblock, 0, stream>>>(Xb, Wqb, bq, Qw, depth_scale, 0);
    proj_kernel<<<pgrid, pblock, 0, stream>>>(Mb, Wkb, bk, Kw, 1.0f,        0);
    proj_kernel<<<pgrid, pblock, 0, stream>>>(Mb, Wvb, bv, Vw, 1.0f,        1);

    dim3 agrid(SEQL / 16, BATCH * NHEAD);             // (128, 64)
    attn_kernel<<<agrid, 32, 0, stream>>>(Qw, Kw, Vw, out);
}
